// MlroleNode_64716567216639
// MI455X (gfx1250) — matrix operand order repaired and checked by reading, not compiled, not run on hardware
//
#include <hip/hip_runtime.h>
#include <hip/hip_bf16.h>

typedef _Float16 v16h __attribute__((ext_vector_type(16)));
typedef float    v8f  __attribute__((ext_vector_type(8)));

#define H      64
#define HEADS  4
#define HID    64
#define RT     4
#define APT    3
#define N_AMB  1023
#define NTOT   1024
#define F_OUT  256      // HEADS*HID
#define SLOPE  0.2f

__device__ __forceinline__ float leaky(float x) { return x >= 0.f ? x : SLOPE * x; }

// ISA 7.12.2: 16-bit A-matrix 16x32 layout. Relative K inside a 32-wide window
// for half-index hi (0..15) of the per-lane v16h fragment.
__device__ __forceinline__ int a_kmap(int lane, int hi) {
    int v = hi >> 1, p = hi & 1;
    return ((v >> 2) << 4) + ((lane >> 4) << 3) + ((v & 3) << 1) + p;
}
// 16-bit B-matrix 32x16: N = lane&15, K = 16*(hi>>3) + 8*(lane>>4) + (hi&7)
__device__ __forceinline__ int b_kmap(int lane, int hi) {
    return ((hi >> 3) << 4) + ((lane >> 4) << 3) + (hi & 7);
}

// ---------------------------------------------------------------------------
// Kernel 1: sequential routing/merge prelude -> hidden_1 (64 floats)
// ---------------------------------------------------------------------------
__global__ void prelude_kernel(const float* __restrict__ hidden,
                               const float* __restrict__ type_agents,
                               const float* __restrict__ W_self,  const float* __restrict__ b_self,
                               const float* __restrict__ W_merge, const float* __restrict__ b_merge,
                               const float* __restrict__ W_trans, const float* __restrict__ b_trans,
                               float* __restrict__ hidden1_out) {
    __shared__ float cat[2 * H];
    const int o = threadIdx.x;                       // 0..63
    float acc = b_self[o];
    #pragma unroll 8
    for (int h = 0; h < H; ++h) acc += hidden[h] * W_self[o * H + h];
    cat[o] = acc;
    __syncthreads();
    for (int t = 0; t < RT; ++t) {
        float tmp = b_trans[t * H + o];
        for (int a = 0; a < APT; ++a) {
            const float* ag = type_agents + (t * APT + a) * H;
            const float* wr = W_trans + (t * H + o) * H;
            float s = 0.f;
            #pragma unroll 8
            for (int h = 0; h < H; ++h) s += ag[h] * wr[h];
            tmp += s * (1.f / 3.f);
        }
        cat[H + o] = tmp;
        __syncthreads();
        float m = b_merge[o];
        #pragma unroll 8
        for (int c = 0; c < 2 * H; ++c) m += cat[c] * W_merge[o * 2 * H + c];
        m = leaky(m);
        __syncthreads();                             // all reads done before overwrite
        cat[o] = m;
        __syncthreads();
    }
    hidden1_out[o] = cat[o];
}

// ---------------------------------------------------------------------------
// Kernel 2: g_{l,r} = h @ W^T   ([1024x64] @ [64x256]) via v_wmma_f32_16x16x32_f16
// grid (16 row-blocks, 2 = {W_l, W_r}), 256 threads = 8 waves, 1 tile / wave-iter
// ---------------------------------------------------------------------------
__global__ void proj_kernel(const float* __restrict__ hidden1,
                            const float* __restrict__ ambiguous,
                            const float* __restrict__ W_l,
                            const float* __restrict__ W_r,
                            float* __restrict__ g_out /* [2][1024][256] */) {
    __shared__ float hb[64 * 65];                    // padded stride: conflict-free column reads
    const int tid  = threadIdx.x;
    const int base = blockIdx.x * 64;
    const float* W = (blockIdx.y == 0) ? W_l : W_r;
    float* g = g_out + (size_t)blockIdx.y * NTOT * F_OUT;

    for (int idx = tid; idx < 64 * 64; idx += 256) { // coalesced stage of 64 h-rows
        int r = idx >> 6, k = idx & 63, grow = base + r;
        hb[r * 65 + k] = (grow == 0) ? hidden1[k] : ambiguous[(size_t)(grow - 1) * H + k];
    }
    __syncthreads();

    const int wave = tid >> 5, lane = tid & 31;
    for (int tile = wave; tile < 64; tile += 8) {    // 4 M-tiles x 16 N-tiles
        const int mt = tile & 3, nt = tile >> 2;
        const int n    = nt * 16 + (lane & 15);
        const int arow = mt * 16 + (lane & 15);
        v8f acc = {};
        #pragma unroll
        for (int kb = 0; kb < H; kb += 32) {
            v16h a, b;
            #pragma unroll
            for (int hi = 0; hi < 16; ++hi) {
                a[hi] = (_Float16)hb[arow * 65 + kb + a_kmap(lane, hi)];
                b[hi] = (_Float16)W[n * H + kb + b_kmap(lane, hi)];
            }
            acc = __builtin_amdgcn_wmma_f32_16x16x32_f16(false, a, false, b,
                                                         (short)0, acc, false, false);
        }
        const int rit = (lane >> 4) << 3;            // D: M = v + 8*(lane>=16)
        #pragma unroll
        for (int v = 0; v < 8; ++v)
            g[(size_t)(base + mt * 16 + rit + v) * F_OUT + nt * 16 + (lane & 15)] = acc[v];
    }
}

// ---------------------------------------------------------------------------
// Kernel 3: only row 0 of the attention is consumed -> single-query GATv2.
// logits + softmax + weighted sum of g_r + head mean -> hidden_2 (64 floats)
// ---------------------------------------------------------------------------
__global__ void attn_kernel(const float* __restrict__ g_buf,   // [2][1024][256]
                            const float* __restrict__ w_attn,
                            float* __restrict__ hidden2_out) {
    const float* g_l = g_buf;
    const float* g_r = g_buf + (size_t)NTOT * F_OUT;
    __shared__ float e_s[HEADS][NTOT];
    __shared__ float gr0[F_OUT];
    __shared__ float red[HEADS][64];
    __shared__ float mh[HEADS], sh[HEADS];
    __shared__ float ar[HEADS][HID];
    const int tid = threadIdx.x;                     // 256 threads

    gr0[tid] = g_r[tid];                             // g_r row 0
    __syncthreads();

    // e[0,j,hd] = sum_f leaky(g_l[j,hd,f] + g_r[0,hd,f]) * w_attn[f]
    for (int idx = tid; idx < NTOT * HEADS; idx += 256) {
        const int j = idx >> 2, hd = idx & 3;
        const float* gl = g_l + (size_t)j * F_OUT + hd * HID;
        const float* g0 = &gr0[hd * HID];
        float acc = 0.f;
        #pragma unroll 8
        for (int f = 0; f < HID; ++f) acc += leaky(gl[f] + g0[f]) * w_attn[f];
        e_s[hd][j] = acc;
    }
    __syncthreads();

    const int hd = tid >> 6, t = tid & 63;           // 64 threads per head
    float lmax = -1e30f;
    for (int j = t; j < NTOT; j += 64) lmax = fmaxf(lmax, e_s[hd][j]);
    red[hd][t] = lmax;
    __syncthreads();
    if (t == 0) { float m = -1e30f; for (int k = 0; k < 64; ++k) m = fmaxf(m, red[hd][k]); mh[hd] = m; }
    __syncthreads();
    const float m = mh[hd];
    float ps = 0.f;
    for (int j = t; j < NTOT; j += 64) { float v = __expf(e_s[hd][j] - m); e_s[hd][j] = v; ps += v; }
    red[hd][t] = ps;
    __syncthreads();
    if (t == 0) { float s = 0.f; for (int k = 0; k < 64; ++k) s += red[hd][k]; sh[hd] = s; }
    __syncthreads();

    // attn_res[0,hd,f] = (1/sh) * sum_j e_s[hd][j] * g_r[j,hd,f]; f = t (coalesced)
    float acc = 0.f;
    for (int j = 0; j < NTOT; ++j) acc += e_s[hd][j] * g_r[(size_t)j * F_OUT + hd * HID + t];
    ar[hd][t] = acc / sh[hd];
    __syncthreads();
    if (tid < HID)
        hidden2_out[tid] = 0.25f * (ar[0][tid] + ar[1][tid] + ar[2][tid] + ar[3][tid]);
}

// ---------------------------------------------------------------------------
// Kernel 4: tail MLP on 1023 rows: [amb|h2](128) ->64 ->128 ->4, sigmoid.
// Layers 1-2 via WMMA; 64 rows per block; xbuf reused for x2 (LDS < 50KB).
// ---------------------------------------------------------------------------
__global__ void mlp_kernel(const float* __restrict__ ambiguous,
                           const float* __restrict__ hidden2,
                           const float* __restrict__ Wd0, const float* __restrict__ bd0,
                           const float* __restrict__ Wd1, const float* __restrict__ bd1,
                           const float* __restrict__ Wd2, const float* __restrict__ bd2,
                           float* __restrict__ out) {
    __shared__ float xbuf[64 * 130];                 // x (64x128), later x2 (64x128)
    __shared__ float x1[64 * 65];                    // 64x64
    const int tid  = threadIdx.x;
    const int base = blockIdx.x * 64;

    for (int idx = tid; idx < 64 * 128; idx += 256) {
        const int r = idx >> 7, k = idx & 127, i = base + r;
        float v;
        if (k < H) v = (i < N_AMB) ? ambiguous[(size_t)i * H + k] : 0.f;
        else       v = hidden2[k - H];
        xbuf[r * 130 + k] = v;
    }
    __syncthreads();

    const int wave = tid >> 5, lane = tid & 31;
    // layer 1: [64x128] @ Wd0^T -> [64x64], +bias, leaky
    for (int tile = wave; tile < 16; tile += 8) {
        const int mt = tile & 3, nt = tile >> 2;
        const int n = nt * 16 + (lane & 15), arow = mt * 16 + (lane & 15);
        v8f acc = {};
        #pragma unroll
        for (int kb = 0; kb < 128; kb += 32) {
            v16h a, b;
            #pragma unroll
            for (int hi = 0; hi < 16; ++hi) {
                a[hi] = (_Float16)xbuf[arow * 130 + kb + a_kmap(lane, hi)];
                b[hi] = (_Float16)Wd0[n * 128 + kb + b_kmap(lane, hi)];
            }
            acc = __builtin_amdgcn_wmma_f32_16x16x32_f16(false, a, false, b,
                                                         (short)0, acc, false, false);
        }
        const int rit = (lane >> 4) << 3;
        const float bias = bd0[n];
        #pragma unroll
        for (int v = 0; v < 8; ++v)
            x1[(mt * 16 + rit + v) * 65 + n] = leaky(acc[v] + bias);
    }
    __syncthreads();
    // layer 2: [64x64] @ Wd1^T -> [64x128], +bias, leaky (overwrites xbuf)
    for (int tile = wave; tile < 32; tile += 8) {
        const int mt = tile & 3, nt = tile >> 2;     // nt 0..7
        const int n = nt * 16 + (lane & 15), arow = mt * 16 + (lane & 15);
        v8f acc = {};
        #pragma unroll
        for (int kb = 0; kb < 64; kb += 32) {
            v16h a, b;
            #pragma unroll
            for (int hi = 0; hi < 16; ++hi) {
                a[hi] = (_Float16)x1[arow * 65 + kb + a_kmap(lane, hi)];
                b[hi] = (_Float16)Wd1[n * 64 + kb + b_kmap(lane, hi)];
            }
            acc = __builtin_amdgcn_wmma_f32_16x16x32_f16(false, a, false, b,
                                                         (short)0, acc, false, false);
        }
        const int rit = (lane >> 4) << 3;
        const float bias = bd1[n];
        #pragma unroll
        for (int v = 0; v < 8; ++v)
            xbuf[(mt * 16 + rit + v) * 130 + n] = leaky(acc[v] + bias);
    }
    __syncthreads();
    // layer 3: 128 -> 4, sigmoid; one (row, class) per thread
    {
        const int r = tid >> 2, c = tid & 3;
        float acc = bd2[c];
        #pragma unroll 8
        for (int k = 0; k < 128; ++k) acc += xbuf[r * 130 + k] * Wd2[c * 128 + k];
        const int i = base + r;
        if (i < N_AMB) out[i * RT + c] = 1.f / (1.f + __expf(-acc));
    }
}

// ---------------------------------------------------------------------------
extern "C" void kernel_launch(void* const* d_in, const int* in_sizes, int n_in,
                              void* d_out, int out_size, void* d_ws, size_t ws_size,
                              hipStream_t stream) {
    const float* hidden    = (const float*)d_in[0];
    const float* ambiguous = (const float*)d_in[1];
    const float* type_ag   = (const float*)d_in[2];
    const float* W_self    = (const float*)d_in[3];
    const float* b_self    = (const float*)d_in[4];
    const float* W_merge   = (const float*)d_in[5];
    const float* b_merge   = (const float*)d_in[6];
    const float* W_trans   = (const float*)d_in[7];
    const float* b_trans   = (const float*)d_in[8];
    const float* W_l       = (const float*)d_in[9];
    const float* W_r       = (const float*)d_in[10];
    const float* w_attn    = (const float*)d_in[11];
    const float* Wd0       = (const float*)d_in[12];
    const float* bd0       = (const float*)d_in[13];
    const float* Wd1       = (const float*)d_in[14];
    const float* bd1       = (const float*)d_in[15];
    const float* Wd2       = (const float*)d_in[16];
    const float* bd2       = (const float*)d_in[17];
    float* out = (float*)d_out;

    // scratch: [hidden1:64][hidden2:64][g_l:1024*256][g_r:1024*256] (~2.1 MB)
    float* ws      = (float*)d_ws;
    float* hidden1 = ws;
    float* hidden2 = ws + 64;
    float* gbuf    = ws + 128;

    prelude_kernel<<<1, 64, 0, stream>>>(hidden, type_ag, W_self, b_self,
                                         W_merge, b_merge, W_trans, b_trans, hidden1);
    proj_kernel<<<dim3(16, 2), 256, 0, stream>>>(hidden1, ambiguous, W_l, W_r, gbuf);
    attn_kernel<<<1, 256, 0, stream>>>(gbuf, w_attn, hidden2);
    mlp_kernel<<<16, 256, 0, stream>>>(ambiguous, hidden2,
                                       Wd0, bd0, Wd1, bd1, Wd2, bd2, out);
}
